// DiscreteMarkovDynamics_33732673142977
// MI455X (gfx1250) — matrix operand order repaired and checked by reading, not compiled, not run on hardware
//
#include <hip/hip_runtime.h>
#include <hip/hip_bf16.h>
#include <stdint.h>

// ---------------------------------------------------------------------------
// DiscreteMarkovDynamics on gfx1250 (MI455X).
// B=32, L=4096 -> 131072 positions. VOCAB=512, EMB=256, HID=64, steps=5.
// Strategy: per step, one wave32 per 16-position tile.
//   GEMM1 (16x256 @ 256x64, bf16 WMMA)  -> relu -> LDS restripe ->
//   GEMM2 (16x64 @ 64x512, bf16 WMMA) fused with Gumbel-max categorical
//   sampling + acceptance test.  x ping-pongs through workspace, 5 launches.
// Weights / embedding table converted+packed to WMMA fragment layout once per
// kernel_launch by small prep kernels.
// ---------------------------------------------------------------------------

#define NPOS   131072   // 32*4096
#define VOCAB  512
#define EMB    256
#define HID    64
#define STEPS  5
#define DT     (0.05f / 5.0f)

typedef __attribute__((ext_vector_type(16))) __bf16 bf16x16;
typedef __attribute__((ext_vector_type(8)))  __bf16 bf16x8;
typedef __attribute__((ext_vector_type(8)))  float  f32x8;

union BF16Frag { bf16x16 v; bf16x8 h[2]; __bf16 e[16]; };

static __device__ __forceinline__ __bf16 f2bf(float f) {
    unsigned u = __float_as_uint(f);
    u += 0x7FFFu + ((u >> 16) & 1u);           // round-to-nearest-even
    unsigned short s = (unsigned short)(u >> 16);
    __bf16 r;
    __builtin_memcpy(&r, &s, 2);
    return r;
}

static __device__ __forceinline__ unsigned mix32(unsigned x) {
    x ^= x >> 16; x *= 0x7feb352du;
    x ^= x >> 15; x *= 0x846ca68bu;
    x ^= x >> 16;
    return x;
}

// ------------------------- prep kernels ------------------------------------

__global__ void cvt_emb_bf16(const float* __restrict__ src,
                             __bf16* __restrict__ dst, int n) {
    int i = blockIdx.x * blockDim.x + threadIdx.x;
    if (i < n) dst[i] = f2bf(src[i]);
}

// W1 [EMB x HID] -> fragments f = kb*4 + nt (kb in 0..7, nt in 0..3)
// packed lane-major: dst[f*512 + lane*16 + j]
__global__ void pack_w1(const float* __restrict__ W1, __bf16* __restrict__ dst) {
    int t = blockIdx.x * blockDim.x + threadIdx.x;   // 0 .. 16383
    if (t >= 8 * 4 * 512) return;
    int f = t >> 9, e = t & 511;
    int lane = e >> 4, j = e & 15;
    int kb = f >> 2, nt = f & 3;
    int k = kb * 32 + 8 * (lane >> 4) + (j & 7) + 16 * (j >> 3);
    int n = nt * 16 + (lane & 15);
    dst[t] = f2bf(W1[k * HID + n]);
}

// W2 [HID x VOCAB] -> fragments f = kb*32 + vt (kb in 0..1, vt in 0..31)
__global__ void pack_w2(const float* __restrict__ W2, __bf16* __restrict__ dst) {
    int t = blockIdx.x * blockDim.x + threadIdx.x;   // 0 .. 32767
    if (t >= 2 * 32 * 512) return;
    int f = t >> 9, e = t & 511;
    int lane = e >> 4, j = e & 15;
    int kb = f >> 5, vt = f & 31;
    int k = kb * 32 + 8 * (lane >> 4) + (j & 7) + 16 * (j >> 3);
    int n = vt * 16 + (lane & 15);
    dst[t] = f2bf(W2[k * VOCAB + n]);
}

// ------------------------- main step kernel --------------------------------

__global__ void __launch_bounds__(128)
markov_step(const int* __restrict__ xin, int* __restrict__ xout,
            const __bf16* __restrict__ embB,
            const __bf16* __restrict__ pW1,
            const __bf16* __restrict__ pW2,
            const float* __restrict__ b1,
            const float* __restrict__ b2,
            int stepIdx) {
    __shared__ float hbufAll[4][16 * 68];          // padded stride 68: no bank conflicts

    const int lane = threadIdx.x & 31;
    const int wid  = threadIdx.x >> 5;
    const int tile = blockIdx.x * 4 + wid;
    const int tilebase = tile * 16;
    const int hh  = lane >> 4;                     // half-wave id
    const int col = lane & 15;
    float* hbuf = hbufAll[wid];

    // current states for the 8 rows this half covers (rows r + 8*hh)
    int xr[8];
#pragma unroll
    for (int r = 0; r < 8; ++r) xr[r] = xin[tilebase + r + 8 * hh];

    // -------- gather A (16x256 bf16) in ISA A-fragment layout --------------
    // lane holds row m = col; frag kb element j<8 -> K = kb*32 + 8*hh + j,
    // element j>=8 -> K = kb*32 + 16 + 8*hh + (j-8).
    const int xm = xin[tilebase + col];
    const __bf16* arow = embB + (size_t)xm * EMB;
    BF16Frag a1[8];
#pragma unroll
    for (int kb = 0; kb < 8; ++kb) {
        int base = kb * 32 + 8 * hh;
        a1[kb].h[0] = *(const bf16x8*)(arow + base);
        a1[kb].h[1] = *(const bf16x8*)(arow + base + 16);
    }

    // -------- GEMM1: h = relu(A @ W1 + b1), write to LDS -------------------
#pragma unroll
    for (int nt = 0; nt < 4; ++nt) {
        float bv = b1[nt * 16 + col];
        f32x8 acc = {bv, bv, bv, bv, bv, bv, bv, bv};
#pragma unroll
        for (int kb = 0; kb < 8; ++kb) {
            bf16x16 bfrag = *(const bf16x16*)(pW1 + (size_t)(kb * 4 + nt) * 512 + lane * 16);
            acc = __builtin_amdgcn_wmma_f32_16x16x32_bf16(
                false, a1[kb].v, false, bfrag, (short)0, acc, false, false);
        }
#pragma unroll
        for (int r = 0; r < 8; ++r) {
            float v = acc[r];
            v = v > 0.0f ? v : 0.0f;
            hbuf[(r + 8 * hh) * 68 + nt * 16 + col] = v;
        }
    }

    // -------- restripe h (D layout) -> A2 fragments (A layout) -------------
    BF16Frag a2[2];
#pragma unroll
    for (int kb = 0; kb < 2; ++kb) {
        int base = kb * 32 + 8 * hh;
#pragma unroll
        for (int j = 0; j < 8; ++j) {
            a2[kb].e[j]     = f2bf(hbuf[col * 68 + base + j]);
            a2[kb].e[j + 8] = f2bf(hbuf[col * 68 + base + 16 + j]);
        }
    }

    // -------- GEMM2 fused with Gumbel-max sampling -------------------------
    float bestKey[8], bestVal[8];
    int   bestIdx[8];
#pragma unroll
    for (int r = 0; r < 8; ++r) {
        bestKey[r] = -__builtin_inff();
        bestVal[r] = 0.0f;
        bestIdx[r] = xr[r];
    }

    const unsigned seed = 0x9E3779B9u * (unsigned)(stepIdx + 1);

    for (int vt = 0; vt < 32; ++vt) {
        // prefetch next vocab tile's fragments into near caches
        __builtin_prefetch(pW2 + (size_t)((vt + 1) & 31) * 512 + lane * 16, 0, 3);
        __builtin_prefetch(pW2 + (size_t)(32 + ((vt + 1) & 31)) * 512 + lane * 16, 0, 3);

        float b2v = b2[vt * 16 + col];
        f32x8 acc = {b2v, b2v, b2v, b2v, b2v, b2v, b2v, b2v};
        bf16x16 w0 = *(const bf16x16*)(pW2 + (size_t)vt * 512 + lane * 16);
        bf16x16 w1 = *(const bf16x16*)(pW2 + (size_t)(32 + vt) * 512 + lane * 16);
        acc = __builtin_amdgcn_wmma_f32_16x16x32_bf16(
            false, a2[0].v, false, w0, (short)0, acc, false, false);
        acc = __builtin_amdgcn_wmma_f32_16x16x32_bf16(
            false, a2[1].v, false, w1, (short)0, acc, false, false);

        const int vcol = vt * 16 + col;
#pragma unroll
        for (int r = 0; r < 8; ++r) {
            int p = tilebase + r + 8 * hh;
            unsigned hsh = mix32((unsigned)p * 0x01000193u ^
                                 (unsigned)vcol * 0x85EBCA6Bu ^ seed);
            float u = (float)(hsh >> 9) * (1.0f / 8388608.0f) + 1e-7f;  // (0,1)
            float g = -__logf(-__logf(u));                              // Gumbel(0,1)
            float key = acc[r] + g;
            if (vcol == xr[r]) key = -__builtin_inff();                 // mask self
            if (key > bestKey[r]) {
                bestKey[r] = key;
                bestVal[r] = acc[r];                                    // rate (intensity)
                bestIdx[r] = vcol;
            }
        }
    }

    // -------- per-row argmax across the 16 lanes of each half --------------
#pragma unroll
    for (int r = 0; r < 8; ++r) {
        float k = bestKey[r], v = bestVal[r];
        int   i = bestIdx[r];
#pragma unroll
        for (int off = 1; off < 16; off <<= 1) {
            float ok = __shfl_xor(k, off, 32);
            float ov = __shfl_xor(v, off, 32);
            int   oi = __shfl_xor(i, off, 32);
            if (ok > k) { k = ok; v = ov; i = oi; }
        }
        if (col == r) {   // lanes 0..7 write rows 0..7, lanes 16..23 rows 8..15
            int p = tilebase + r + 8 * hh;
            unsigned h2 = mix32((unsigned)p * 0x27d4eb2du ^ seed ^ 0xA5A5A5A5u);
            float u2 = (float)(h2 >> 9) * (1.0f / 8388608.0f);
            float paccept = 1.0f - __expf(-v * DT);
            xout[p] = (u2 < paccept) ? i : xr[r];
        }
    }
}

// ------------------------- host-side launch --------------------------------

extern "C" void kernel_launch(void* const* d_in, const int* in_sizes, int n_in,
                              void* d_out, int out_size, void* d_ws, size_t ws_size,
                              hipStream_t stream) {
    (void)in_sizes; (void)n_in; (void)out_size; (void)ws_size;

    const int*   x    = (const int*)  d_in[0];   // [32,4096] int32
    const float* emb  = (const float*)d_in[1];   // [512,256]
    const float* W1   = (const float*)d_in[2];   // [256,64]
    const float* b1   = (const float*)d_in[3];   // [64]
    const float* W2   = (const float*)d_in[4];   // [64,512]
    const float* b2   = (const float*)d_in[5];   // [512]
    // d_in[6] = steps (device scalar) -- fixed to 5 per reference setup.

    char* ws = (char*)d_ws;
    __bf16* embB = (__bf16*)(ws + 0);            // 512*256*2   = 262144 B
    __bf16* pW1  = (__bf16*)(ws + 262144);       // 16384*2     =  32768 B
    __bf16* pW2  = (__bf16*)(ws + 294912);       // 32768*2     =  65536 B
    int*    xA   = (int*)   (ws + 360448);       // 131072*4    = 524288 B
    int*    xB   = (int*)   (ws + 884736);       // 131072*4    = 524288 B
    int*    xout = (int*)d_out;

    // prep: bf16 conversion + fragment packing (deterministic each call)
    cvt_emb_bf16<<<(VOCAB * EMB + 255) / 256, 256, 0, stream>>>(emb, embB, VOCAB * EMB);
    pack_w1<<<(8 * 4 * 512 + 255) / 256, 256, 0, stream>>>(W1, pW1);
    pack_w2<<<(2 * 32 * 512 + 255) / 256, 256, 0, stream>>>(W2, pW2);

    // 5 steps, ping-pong through workspace, final step writes d_out
    const int blocks = NPOS / 16 / 4;            // 4 waves (tiles) per 128-thread WG
    const int* src = x;
    for (int s = 0; s < STEPS; ++s) {
        int* dst = (s == STEPS - 1) ? xout : ((s & 1) ? xB : xA);
        markov_step<<<blocks, 128, 0, stream>>>(src, dst, embB, pW1, pW2, b1, b2, s);
        src = dst;
    }
}
